// GPT_87428354277600
// MI455X (gfx1250) — compile-verified, matrix operand order checked by reading, not run on hardware
//
#include <hip/hip_runtime.h>
#include <hip/hip_bf16.h>

typedef unsigned short u16;
typedef unsigned int   u32;
typedef __attribute__((ext_vector_type(16))) __bf16 v16bf;
typedef __attribute__((ext_vector_type(8)))  float  v8f;

#define BB 2
#define TT 512
#define CC 768
#define HH 12
#define HDIM 64
#define VV 50304
#define KPAD 800      // 769 padded up to multiple of 32
#define NSTEPS 20
#define HSTEP 0.05f
#define LOGN 6.2383246250395075f   // log(512)

static __device__ __forceinline__ u16 f2bf(float f) {
  u32 x = __float_as_uint(f);
  u32 r = x + 0x7FFFu + ((x >> 16) & 1u);   // round-to-nearest-even
  return (u16)(r >> 16);
}

// ---------- CDNA5 async global->LDS copy (probe via __has_builtin) ----------
#if __has_builtin(__builtin_amdgcn_global_load_async_to_lds_b128)
#define HAS_ASYNC_LDS 1
typedef int v4i_ __attribute__((vector_size(16)));
typedef __attribute__((address_space(1))) v4i_* as1_v4i;
typedef __attribute__((address_space(3))) v4i_* as3_v4i;
static __device__ __forceinline__ void async_cp16(const void* g, void* l) {
  __builtin_amdgcn_global_load_async_to_lds_b128(
      (as1_v4i)(unsigned long long)g,
      (as3_v4i)(unsigned int)(unsigned long long)l, 0, 0);
}
static __device__ __forceinline__ void async_wait0() {
#if __has_builtin(__builtin_amdgcn_s_wait_asynccnt)
  __builtin_amdgcn_s_wait_asynccnt(0);
#else
  asm volatile("s_wait_asynccnt 0x0" ::: "memory");
#endif
}
#else
#define HAS_ASYNC_LDS 0
#endif

// ---------------- embedding: z = wte[idx] + wpe[pos] ----------------
__global__ void embed_kernel(const int* __restrict__ idx, const float* __restrict__ wte,
                             const float* __restrict__ wpe, float* __restrict__ z) {
  long i = (long)blockIdx.x * blockDim.x + threadIdx.x;
  if (i >= (long)BB * TT * CC) return;
  int c = (int)(i % CC);
  long bt = i / CC;
  int t = (int)(bt % TT);
  z[i] = wte[(long)idx[bt] * CC + c] + wpe[(long)t * CC + c];
}

// --- fp32 weight [K,N] -> bf16 TRANSPOSED, zero-padded: dst[n*Kpad+k] = src[k*N+n] ---
__global__ void convert_pad_T_kernel(const float* __restrict__ src, u16* __restrict__ dst,
                                     int K, int N, int Kpad) {
  long i = (long)blockIdx.x * blockDim.x + threadIdx.x;
  if (i >= (long)N * Kpad) return;
  int n = (int)(i / Kpad), k = (int)(i % Kpad);
  dst[i] = (k < K) ? f2bf(src[(long)k * N + n]) : (u16)0;
}

// --- LayerNorm over concat(x[row,0:C], t) -> bf16 [row, KPAD], pad=0 ---
__global__ void ln_concat_kernel(const float* __restrict__ x, float tval,
                                 const float* __restrict__ w, const float* __restrict__ b,
                                 u16* __restrict__ out) {
  __shared__ float red[256];
  __shared__ float s_mu, s_rs;
  int row = blockIdx.x, tid = threadIdx.x;
  const float* xr = x + (long)row * CC;
  float s = 0.f, sq = 0.f;
  for (int c = tid; c < CC; c += 256) { float v = xr[c]; s += v; sq += v * v; }
  if (tid == 0) { s += tval; sq += tval * tval; }
  red[tid] = s; __syncthreads();
  for (int k = 128; k > 0; k >>= 1) { if (tid < k) red[tid] += red[tid + k]; __syncthreads(); }
  float tots = red[0]; __syncthreads();
  red[tid] = sq; __syncthreads();
  for (int k = 128; k > 0; k >>= 1) { if (tid < k) red[tid] += red[tid + k]; __syncthreads(); }
  float totq = red[0];
  if (tid == 0) {
    float mu = tots / (float)(CC + 1);
    float var = totq / (float)(CC + 1) - mu * mu;
    s_mu = mu; s_rs = rsqrtf(var + 1e-5f);
  }
  __syncthreads();
  float mu = s_mu, rs = s_rs;
  u16* orow = out + (long)row * KPAD;
  for (int c = tid; c < KPAD; c += 256) {
    float v;
    if (c < CC)       v = (xr[c] - mu) * rs * w[c] + b[c];
    else if (c == CC) v = (tval - mu) * rs * w[CC] + b[CC];
    else              v = 0.f;
    orow[c] = f2bf(v);
  }
}

// ---- split qkv bf16 [B*T,3C] -> q[B,H,T,64], k[B,H,T,64] (BT for scores), vT[B,H,64,T] ----
__global__ void qkv_split_kernel(const u16* __restrict__ qkv, u16* __restrict__ q,
                                 u16* __restrict__ kb, u16* __restrict__ vT) {
  long i = (long)blockIdx.x * blockDim.x + threadIdx.x;
  if (i >= (long)BB * TT * 3 * CC) return;
  int c3 = (int)(i % (3 * CC));
  long bt = i / (3 * CC);
  int b = (int)(bt / TT), t = (int)(bt % TT);
  int which = c3 / CC, c = c3 % CC;
  int h = c / HDIM, d = c % HDIM;
  u16 val = qkv[i];
  long bh = (long)b * HH + h;
  if (which == 0)      q[(bh * TT + t) * HDIM + d] = val;
  else if (which == 1) kb[(bh * TT + t) * HDIM + d] = val;
  else                 vT[(bh * HDIM + d) * TT + t] = val;
}

// ----------- causal masked softmax, in-place on att [B*H*T rows] -----------
__global__ void softmax_mask_kernel(float* __restrict__ att) {
  __shared__ float red[256];
  int r = blockIdx.x, tid = threadIdx.x;
  int i = r % TT;
  float* row = att + (long)r * TT;
  float m = -1e30f;
  for (int j = tid; j < TT; j += 256) if (j <= i) m = fmaxf(m, row[j]);
  red[tid] = m; __syncthreads();
  for (int k = 128; k > 0; k >>= 1) { if (tid < k) red[tid] = fmaxf(red[tid], red[tid + k]); __syncthreads(); }
  m = red[0]; __syncthreads();
  float s = 0.f;
  for (int j = tid; j < TT; j += 256) if (j <= i) s += expf(row[j] - m);
  red[tid] = s; __syncthreads();
  for (int k = 128; k > 0; k >>= 1) { if (tid < k) red[tid] += red[tid + k]; __syncthreads(); }
  s = red[0];
  float inv = 1.0f / s;
  for (int j = tid; j < TT; j += 256) row[j] = (j <= i) ? expf(row[j] - m) * inv : 0.f;
}

// =============== Sinkhorn (log-domain, eps=1), split per half-iteration ===============
__global__ void sinkhorn_init(float* __restrict__ su, float* __restrict__ sv) {
  int i = blockIdx.x * 256 + threadIdx.x;
  if (i < BB * HH * TT) { su[i] = 0.f; sv[i] = 0.f; }
}

// u_i = -log(n) - lse_j(v_j - C_ij); one wave per row, streaming lse
__global__ __launch_bounds__(256) void sinkhorn_u(const float* __restrict__ Cmat,
                                                  const float* __restrict__ sv,
                                                  float* __restrict__ su) {
  int gw = (blockIdx.x * 256 + threadIdx.x) >> 5;  // global wave id in [0, 24*512)
  int lane = threadIdx.x & 31;
  int bh = gw >> 9, i = gw & 511;
  const float* Cr = Cmat + ((long)bh * TT + i) * TT;
  const float* v = sv + (long)bh * TT;
  float m = -1e30f, s = 0.f;
  for (int j = lane; j < TT; j += 32) {
    float x = v[j] - Cr[j];
    float mn = fmaxf(m, x);
    s = s * expf(m - mn) + expf(x - mn);
    m = mn;
  }
#pragma unroll
  for (int o = 16; o > 0; o >>= 1) {
    float mo = __shfl_xor(m, o, 32);
    float so = __shfl_xor(s, o, 32);
    float mn = fmaxf(m, mo);
    s = s * expf(m - mn) + so * expf(mo - mn);
    m = mn;
  }
  if (lane == 0) su[(long)bh * TT + i] = -LOGN - (m + logf(s));
}

// v_j = -log(n) - lse_i(u_i - C_ij); thread per column, coalesced streaming
__global__ __launch_bounds__(256) void sinkhorn_v(const float* __restrict__ Cmat,
                                                  const float* __restrict__ su,
                                                  float* __restrict__ sv) {
  int bh = blockIdx.y;
  int j = blockIdx.x * 256 + threadIdx.x;
  const float* Cb = Cmat + (long)bh * TT * TT;
  const float* u = su + (long)bh * TT;
  float m = -1e30f, s = 0.f;
  for (int i = 0; i < TT; ++i) {
    float x = u[i] - Cb[(long)i * TT + j];
    float mn = fmaxf(m, x);
    s = s * expf(m - mn) + expf(x - mn);
    m = mn;
  }
  sv[(long)bh * TT + j] = -LOGN - (m + logf(s));
}

// pi = exp(u_i + v_j - C_ij) * n  -> bf16
__global__ void sinkhorn_pi_kernel(const float* __restrict__ Cmat, const float* __restrict__ su,
                                   const float* __restrict__ sv, u16* __restrict__ pi) {
  long idx = (long)blockIdx.x * 256 + threadIdx.x;
  if (idx >= (long)BB * HH * TT * TT) return;
  int j = (int)(idx & (TT - 1));
  long r = idx >> 9;
  int i = (int)(r & (TT - 1));
  int bh = (int)(r >> 9);
  float val = expf(su[(long)bh * TT + i] + sv[(long)bh * TT + j] - Cmat[idx]) * (float)TT;
  pi[idx] = f2bf(val);
}

// ---------------- z += a * dz ----------------
__global__ void axpy_kernel(float* __restrict__ z, const float* __restrict__ dz, float a, long n) {
  long i = (long)blockIdx.x * blockDim.x + threadIdx.x;
  if (i < n) z[i] += a * dz[i];
}

// ---------------- final LN on last token of each batch ----------------
__global__ void lnf_last_kernel(const float* __restrict__ z, const float* __restrict__ w,
                                const float* __restrict__ b, float* __restrict__ xf) {
  __shared__ float red[256];
  __shared__ float s_mu, s_rs;
  int bi = blockIdx.x, tid = threadIdx.x;
  const float* xr = z + ((long)(bi + 1) * TT - 1) * CC;
  float s = 0.f, sq = 0.f;
  for (int c = tid; c < CC; c += 256) { float v = xr[c]; s += v; sq += v * v; }
  red[tid] = s; __syncthreads();
  for (int k = 128; k > 0; k >>= 1) { if (tid < k) red[tid] += red[tid + k]; __syncthreads(); }
  float tots = red[0]; __syncthreads();
  red[tid] = sq; __syncthreads();
  for (int k = 128; k > 0; k >>= 1) { if (tid < k) red[tid] += red[tid + k]; __syncthreads(); }
  float totq = red[0];
  if (tid == 0) {
    float mu = tots / (float)CC;
    s_mu = mu; s_rs = rsqrtf(totq / (float)CC - mu * mu + 1e-5f);
  }
  __syncthreads();
  float mu = s_mu, rs = s_rs;
  for (int c = tid; c < CC; c += 256)
    xf[(long)bi * CC + c] = (xr[c] - mu) * rs * w[c] + b[c];
}

// ---------------- logits = xf @ wte^T (bandwidth-bound, M=2) ----------------
__global__ void logits_kernel(const float* __restrict__ xf, const float* __restrict__ wte,
                              float* __restrict__ out) {
  int v = blockIdx.x * blockDim.x + threadIdx.x;
  if (v >= VV) return;
  const float* wr = wte + (long)v * CC;
  float s0 = 0.f, s1 = 0.f;
  for (int c = 0; c < CC; c += 4) {
    float4 w4 = *(const float4*)(wr + c);
    float4 a4 = *(const float4*)(xf + c);
    float4 b4 = *(const float4*)(xf + CC + c);
    s0 += w4.x * a4.x + w4.y * a4.y + w4.z * a4.z + w4.w * a4.w;
    s1 += w4.x * b4.x + w4.y * b4.y + w4.z * b4.z + w4.w * b4.w;
  }
  out[v] = s0;
  out[(long)VV + v] = s1;
}

// =================== batched bf16 WMMA GEMM (B given COL-MAJOR: BT[N][K]) ===================
// Tile 128x64, BK=32, 128 threads (4 waves), each wave: 32x64 = 8 WMMA frags.
// Double-buffered LDS; A and BT tiles are identical 16B block copies (no transpose VALU).
// M mult of 128, N mult of 64, K mult of 32. Batch offset: (z/Hdiv)*xOut + (z%Hdiv)*xIn.
union AF { v16bf v; uint4 q[2]; };
union CF { v8f v; float f[8]; };

template <bool OUTF, bool OUTB, bool HASBIAS, bool GELU>
__global__ __launch_bounds__(128, 1) void gemm_wmma_bf16(
    const u16* __restrict__ A, int lda, long aOut, long aIn,
    const u16* __restrict__ BT, int ldbt, long bOut, long bIn,
    float* __restrict__ outF, u16* __restrict__ outB, int ldc, long cOut, long cIn,
    const float* __restrict__ bias, int K, float scale, int Hdiv) {
  __shared__ u16 As[2][128][40];   // row stride 80B: 16B-aligned subrows
  __shared__ u16 Bs[2][64][40];    // n-major
  int z = blockIdx.z;
  long aoff = (long)(z / Hdiv) * aOut + (long)(z % Hdiv) * aIn;
  long boff = (long)(z / Hdiv) * bOut + (long)(z % Hdiv) * bIn;
  long coff = (long)(z / Hdiv) * cOut + (long)(z % Hdiv) * cIn;
  int tid = threadIdx.x;
  int wave = tid >> 5, lane = tid & 31;
  int half = lane >> 4, l16 = lane & 15;
  int koff = half * 8;            // lanes 0-15: K {0..7,16..23}; 16-31: {8..15,24..31}
  int blockM = blockIdx.y * 128, blockN = blockIdx.x * 64;

  int grow_ = tid >> 2;           // 0..31 (+32/64/96 for A; +32 for B)
  int gcol_ = (tid & 3) * 8;      // 0,8,16,24

  const u16* Aptr = A + aoff + (long)(blockM + grow_) * lda + gcol_;
  const u16* Bptr = BT + boff + (long)(blockN + grow_) * ldbt + gcol_;

#if !HAS_ASYNC_LDS
  uint4 ra[4], rb[2];
#endif

  auto issue_tile = [&](int kt, int buf) {
#if HAS_ASYNC_LDS
    async_cp16(Aptr + kt,                  &As[buf][grow_][gcol_]);
    async_cp16(Aptr + kt + (long)32 * lda, &As[buf][grow_ + 32][gcol_]);
    async_cp16(Aptr + kt + (long)64 * lda, &As[buf][grow_ + 64][gcol_]);
    async_cp16(Aptr + kt + (long)96 * lda, &As[buf][grow_ + 96][gcol_]);
    async_cp16(Bptr + kt,                  &Bs[buf][grow_][gcol_]);
    async_cp16(Bptr + kt + (long)32 * ldbt, &Bs[buf][grow_ + 32][gcol_]);
    (void)buf;
#else
    ra[0] = *(const uint4*)(Aptr + kt);
    ra[1] = *(const uint4*)(Aptr + kt + (long)32 * lda);
    ra[2] = *(const uint4*)(Aptr + kt + (long)64 * lda);
    ra[3] = *(const uint4*)(Aptr + kt + (long)96 * lda);
    rb[0] = *(const uint4*)(Bptr + kt);
    rb[1] = *(const uint4*)(Bptr + kt + (long)32 * ldbt);
#endif
  };
#if !HAS_ASYNC_LDS
  auto store_tile = [&](int buf) {
    *(uint4*)&As[buf][grow_][gcol_] = ra[0];
    *(uint4*)&As[buf][grow_ + 32][gcol_] = ra[1];
    *(uint4*)&As[buf][grow_ + 64][gcol_] = ra[2];
    *(uint4*)&As[buf][grow_ + 96][gcol_] = ra[3];
    *(uint4*)&Bs[buf][grow_][gcol_] = rb[0];
    *(uint4*)&Bs[buf][grow_ + 32][gcol_] = rb[1];
  };
#endif

  v8f acc[2][4] = {};
  int ntiles = K >> 5;
  int buf = 0;
  issue_tile(0, 0);
#if HAS_ASYNC_LDS
  async_wait0();
#else
  store_tile(0);
#endif
  __syncthreads();

  for (int t = 0; t < ntiles; ++t) {
    if (t + 1 < ntiles) issue_tile((t + 1) << 5, buf ^ 1);
    // load all 12 fragments up front, then 8 WMMAs
    AF a0, a1, b0, b1, b2, b3;
    a0.q[0] = *(const uint4*)&As[buf][wave * 32 + l16][koff];
    a0.q[1] = *(const uint4*)&As[buf][wave * 32 + l16][koff + 16];
    a1.q[0] = *(const uint4*)&As[buf][wave * 32 + 16 + l16][koff];
    a1.q[1] = *(const uint4*)&As[buf][wave * 32 + 16 + l16][koff + 16];
    b0.q[0] = *(const uint4*)&Bs[buf][l16][koff];
    b0.q[1] = *(const uint4*)&Bs[buf][l16][koff + 16];
    b1.q[0] = *(const uint4*)&Bs[buf][16 + l16][koff];
    b1.q[1] = *(const uint4*)&Bs[buf][16 + l16][koff + 16];
    b2.q[0] = *(const uint4*)&Bs[buf][32 + l16][koff];
    b2.q[1] = *(const uint4*)&Bs[buf][32 + l16][koff + 16];
    b3.q[0] = *(const uint4*)&Bs[buf][48 + l16][koff];
    b3.q[1] = *(const uint4*)&Bs[buf][48 + l16][koff + 16];
    acc[0][0] = __builtin_amdgcn_wmma_f32_16x16x32_bf16(false, a0.v, false, b0.v, (short)0, acc[0][0], false, false);
    acc[1][0] = __builtin_amdgcn_wmma_f32_16x16x32_bf16(false, a1.v, false, b0.v, (short)0, acc[1][0], false, false);
    acc[0][1] = __builtin_amdgcn_wmma_f32_16x16x32_bf16(false, a0.v, false, b1.v, (short)0, acc[0][1], false, false);
    acc[1][1] = __builtin_amdgcn_wmma_f32_16x16x32_bf16(false, a1.v, false, b1.v, (short)0, acc[1][1], false, false);
    acc[0][2] = __builtin_amdgcn_wmma_f32_16x16x32_bf16(false, a0.v, false, b2.v, (short)0, acc[0][2], false, false);
    acc[1][2] = __builtin_amdgcn_wmma_f32_16x16x32_bf16(false, a1.v, false, b2.v, (short)0, acc[1][2], false, false);
    acc[0][3] = __builtin_amdgcn_wmma_f32_16x16x32_bf16(false, a0.v, false, b3.v, (short)0, acc[0][3], false, false);
    acc[1][3] = __builtin_amdgcn_wmma_f32_16x16x32_bf16(false, a1.v, false, b3.v, (short)0, acc[1][3], false, false);
    if (t + 1 < ntiles) {
#if HAS_ASYNC_LDS
      async_wait0();
#else
      store_tile(buf ^ 1);
#endif
      __syncthreads();
      buf ^= 1;
    }
  }

#pragma unroll
  for (int mb = 0; mb < 2; ++mb) {
#pragma unroll
    for (int f = 0; f < 4; ++f) {
      int gcol = blockN + f * 16 + l16;
      float bv = HASBIAS ? bias[gcol] : 0.f;
      CF cf; cf.v = acc[mb][f];
#pragma unroll
      for (int e = 0; e < 8; ++e) {
        int grow = blockM + wave * 32 + mb * 16 + e + half * 8;
        float val = cf.f[e] * scale + bv;
        if (GELU) val = 0.5f * val * (1.0f + erff(val * 0.70710678118f));
        long o = coff + (long)grow * ldc + gcol;
        if (OUTF) outF[o] = val;
        if (OUTB) outB[o] = f2bf(val);
      }
    }
  }
}

extern "C" void kernel_launch(void* const* d_in, const int* in_sizes, int n_in,
                              void* d_out, int out_size, void* d_ws, size_t ws_size,
                              hipStream_t stream) {
  const int*   idx        = (const int*)d_in[0];
  const float* wte        = (const float*)d_in[1];
  const float* wpe        = (const float*)d_in[2];
  const float* attn_ln_w  = (const float*)d_in[3];
  const float* attn_ln_b  = (const float*)d_in[4];
  const float* c_attn_w   = (const float*)d_in[5];
  const float* c_attn_b   = (const float*)d_in[6];
  const float* c_proj_w   = (const float*)d_in[7];
  const float* c_proj_b   = (const float*)d_in[8];
  const float* mlp_ln_w   = (const float*)d_in[9];
  const float* mlp_ln_b   = (const float*)d_in[10];
  const float* c_fc_w     = (const float*)d_in[11];
  const float* c_fc_b     = (const float*)d_in[12];
  const float* mlp_proj_w = (const float*)d_in[13];
  const float* mlp_proj_b = (const float*)d_in[14];
  const float* lnf_w      = (const float*)d_in[15];
  const float* lnf_b      = (const float*)d_in[16];
  float* out = (float*)d_out;

  char* ws = (char*)d_ws;
  size_t off = 0;
  auto alloc = [&](size_t bytes) -> void* {
    void* p = ws + off;
    off = (off + bytes + 255) & ~(size_t)255;
    return p;
  };
  // transposed bf16 weights [N][Kpad]
  u16* wA   = (u16*)alloc((size_t)3 * CC * KPAD * 2);
  u16* wP   = (u16*)alloc((size_t)CC * CC * 2);
  u16* wF   = (u16*)alloc((size_t)4 * CC * KPAD * 2);
  u16* wM   = (u16*)alloc((size_t)CC * 4 * CC * 2);
  float* z  = (float*)alloc((size_t)BB * TT * CC * 4);
  u16* xln  = (u16*)alloc((size_t)BB * TT * KPAD * 2);
  u16* qkvB = (u16*)alloc((size_t)BB * TT * 3 * CC * 2);
  u16* qb   = (u16*)alloc((size_t)BB * HH * TT * HDIM * 2);
  u16* kb   = (u16*)alloc((size_t)BB * HH * TT * HDIM * 2);
  u16* vTb  = (u16*)alloc((size_t)BB * HH * HDIM * TT * 2);
  float* att = (float*)alloc((size_t)BB * HH * TT * TT * 4);
  u16* piB  = (u16*)alloc((size_t)BB * HH * TT * TT * 2);
  u16* ymrg = (u16*)alloc((size_t)BB * TT * CC * 2);
  float* attn_out = (float*)alloc((size_t)BB * TT * CC * 4);
  u16* ln2  = (u16*)alloc((size_t)BB * TT * KPAD * 2);
  u16* h1   = (u16*)alloc((size_t)BB * TT * 4 * CC * 2);
  float* dz = (float*)alloc((size_t)BB * TT * CC * 4);
  float* xf = (float*)alloc((size_t)BB * CC * 4);
  float* su = (float*)alloc((size_t)BB * HH * TT * 4);
  float* sv = (float*)alloc((size_t)BB * HH * TT * 4);

  auto cdiv = [](long a, long b) { return (int)((a + b - 1) / b); };

  convert_pad_T_kernel<<<cdiv((long)3 * CC * KPAD, 256), 256, 0, stream>>>(c_attn_w, wA, CC + 1, 3 * CC, KPAD);
  convert_pad_T_kernel<<<cdiv((long)CC * CC, 256), 256, 0, stream>>>(c_proj_w, wP, CC, CC, CC);
  convert_pad_T_kernel<<<cdiv((long)4 * CC * KPAD, 256), 256, 0, stream>>>(c_fc_w, wF, CC + 1, 4 * CC, KPAD);
  convert_pad_T_kernel<<<cdiv((long)CC * 4 * CC, 256), 256, 0, stream>>>(mlp_proj_w, wM, 4 * CC, CC, 4 * CC);

  embed_kernel<<<cdiv((long)BB * TT * CC, 256), 256, 0, stream>>>(idx, wte, wpe, z);

  const long sQ   = (long)TT * HDIM;   // per-head q/k stride
  const long sVT  = (long)HDIM * TT;   // per-head vT stride
  const long sAtt = (long)TT * TT;

  for (int step = 0; step < NSTEPS; ++step) {
    float tval = (float)step * HSTEP;

    ln_concat_kernel<<<BB * TT, 256, 0, stream>>>(z, tval, attn_ln_w, attn_ln_b, xln);

    // qkv = xln @ c_attn + b   [1024,800]x[800,2304] -> bf16
    dim3 g1(3 * CC / 64, BB * TT / 128, 1);
    gemm_wmma_bf16<false, true, true, false><<<g1, 128, 0, stream>>>(
        xln, KPAD, 0, 0, wA, KPAD, 0, 0,
        nullptr, qkvB, 3 * CC, 0, 0, c_attn_b, KPAD, 1.f, 1);
    qkv_split_kernel<<<cdiv((long)BB * TT * 3 * CC, 256), 256, 0, stream>>>(qkvB, qb, kb, vTb);

    // att = (q @ k^T) / 8  over 24 heads; BT operand = k (natural [T,64] layout)
    dim3 g2(TT / 64, TT / 128, BB * HH);
    gemm_wmma_bf16<true, false, false, false><<<g2, 128, 0, stream>>>(
        qb, HDIM, HH * sQ, sQ, kb, HDIM, HH * sQ, sQ,
        att, nullptr, TT, HH * sAtt, sAtt, nullptr, HDIM, 0.125f, HH);

    softmax_mask_kernel<<<BB * HH * TT, 256, 0, stream>>>(att);

    // Sinkhorn: 6 iterations of (u-pass, v-pass), then pi
    sinkhorn_init<<<cdiv((long)BB * HH * TT, 256), 256, 0, stream>>>(su, sv);
    for (int it = 0; it < 6; ++it) {
      sinkhorn_u<<<BB * HH * TT / 8, 256, 0, stream>>>(att, sv, su);
      dim3 gv(TT / 256, BB * HH);
      sinkhorn_v<<<gv, 256, 0, stream>>>(att, su, sv);
    }
    sinkhorn_pi_kernel<<<cdiv((long)BB * HH * TT * TT, 256), 256, 0, stream>>>(att, su, sv, piB);

    // y = pi @ v -> merged [B,T,C] bf16; BT operand = vT [64,T]
    dim3 g3(HDIM / 64, TT / 128, BB * HH);
    gemm_wmma_bf16<false, true, false, false><<<g3, 128, 0, stream>>>(
        piB, TT, HH * sAtt, sAtt, vTb, TT, HH * sVT, sVT,
        nullptr, ymrg, CC, (long)TT * CC, HDIM, nullptr, TT, 1.f, HH);

    // attn_out = y @ c_proj + b   [1024,768]x[768,768] -> f32
    dim3 g4(CC / 64, BB * TT / 128, 1);
    gemm_wmma_bf16<true, false, true, false><<<g4, 128, 0, stream>>>(
        ymrg, CC, 0, 0, wP, CC, 0, 0,
        attn_out, nullptr, CC, 0, 0, c_proj_b, CC, 1.f, 1);

    ln_concat_kernel<<<BB * TT, 256, 0, stream>>>(attn_out, tval, mlp_ln_w, mlp_ln_b, ln2);

    // h1 = gelu(ln2 @ c_fc + b)   [1024,800]x[800,3072] -> bf16
    dim3 g5(4 * CC / 64, BB * TT / 128, 1);
    gemm_wmma_bf16<false, true, true, true><<<g5, 128, 0, stream>>>(
        ln2, KPAD, 0, 0, wF, KPAD, 0, 0,
        nullptr, h1, 4 * CC, 0, 0, c_fc_b, KPAD, 1.f, 1);

    // dz = h1 @ mlp_proj + b   [1024,3072]x[3072,768] -> f32
    dim3 g6(CC / 64, BB * TT / 128, 1);
    gemm_wmma_bf16<true, false, true, false><<<g6, 128, 0, stream>>>(
        h1, 4 * CC, 0, 0, wM, 4 * CC, 0, 0,
        dz, nullptr, CC, 0, 0, mlp_proj_b, 4 * CC, 1.f, 1);

    axpy_kernel<<<cdiv((long)BB * TT * CC, 256), 256, 0, stream>>>(z, dz, HSTEP, (long)BB * TT * CC);
  }

  lnf_last_kernel<<<BB, 256, 0, stream>>>(z, lnf_w, lnf_b, xf);
  logits_kernel<<<cdiv(VV, 256), 256, 0, stream>>>(xf, wte, out);
}
